// HGATActorCritic_18597208391721
// MI455X (gfx1250) — matrix operand order repaired and checked by reading, not compiled, not run on hardware
//
#include <hip/hip_runtime.h>
#include <hip/hip_bf16.h>

// ---------------------------------------------------------------------------
// CDNA5 (gfx1250) HGT actor-critic. Wave32, WMMA bf16 16x16x32.
// ---------------------------------------------------------------------------

typedef __attribute__((ext_vector_type(16))) __bf16 bf16x16;
typedef __attribute__((ext_vector_type(8)))  float  v8f;

union Frag { uint4 q[2]; bf16x16 v; };

__device__ __forceinline__ unsigned encF(float f) {
    unsigned u = __float_as_uint(f);
    return (u & 0x80000000u) ? ~u : (u | 0x80000000u);
}
__device__ __forceinline__ float decF(unsigned e) {
    unsigned u = (e & 0x80000000u) ? (e & 0x7fffffffu) : ~e;
    return __uint_as_float(u);
}

// --------------------------- weight pre-swizzle ----------------------------
// Pack B [KxN] f32 (mode 0) or block-diag from [4][32][32] (mode 1) into the
// bf16 WMMA B-fragment layout: lane = (n&15) + 16*((k>>3)&1), dword v=((k&7)>>1)+4*((k>>4)&1).
__global__ void pack_b_kernel(const float* __restrict__ W, __bf16* __restrict__ out,
                              int K, int N, int mode) {
    int t = blockIdx.x * 256 + threadIdx.x;
    if (t >= K * N) return;
    int k = t / N, n = t % N;
    float w;
    if (mode == 0) {
        w = W[(size_t)k * N + n];
    } else {
        int hk = k >> 5, hn = n >> 5;
        w = (hk == hn) ? W[((size_t)hk * 32 + (k & 31)) * 32 + (n & 31)] : 0.0f;
    }
    int tileN = n >> 4;
    int lane  = (n & 15) + (((k >> 3) & 1) << 4);
    int v     = ((k & 7) >> 1) + (((k >> 4) & 1) << 2);
    int hi    = k & 1;
    int KT    = K >> 5;
    size_t idx = ((((size_t)tileN * KT + (k >> 5)) * 32 + lane) * 8 + v) * 2 + hi;
    out[idx] = (__bf16)w;
}

// ------------------------------- embeddings --------------------------------
__global__ void embed_kernel(const float* __restrict__ X, const float* __restrict__ W,
                             const float* __restrict__ b, float* __restrict__ outF,
                             __bf16* __restrict__ outB, int N, int Kin) {
    int t = blockIdx.x * 256 + threadIdx.x;
    if (t >= N * 128) return;
    int n = t >> 7, c = t & 127;
    float acc = b[c];
    for (int j = 0; j < Kin; ++j) acc += X[(size_t)n * Kin + j] * W[j * 128 + c];
    outF[t] = acc;
    outB[t] = (__bf16)acc;
}

// --------------------------- generic WMMA GEMM -----------------------------
// C[M,N] = A[M,K](bf16,row-major,lda) @ Bpack + bias; optional learned-skip
// residual (beta=sigmoid(skip)), optional relu; writes f32 and/or bf16.
// Requires: M%16==0, N%128==0 handled by grid, K%32==0. Block=256 (8 waves),
// wave w owns the 16x16 tile at (blockIdx.x*16, blockIdx.y*128 + w*16).
__global__ __launch_bounds__(256) void gemm_wmma(
    const __bf16* __restrict__ A, int lda, const __bf16* __restrict__ Bp,
    const float* __restrict__ bias, const float* __restrict__ resid,
    const float* __restrict__ skipP, float* __restrict__ outF,
    __bf16* __restrict__ outB, int M, int K, int N, int relu) {
    int wave = threadIdx.x >> 5, lane = threadIdx.x & 31;
    int m0 = blockIdx.x << 4;
    int n0 = (blockIdx.y << 7) + (wave << 4);
    if (n0 >= N) return;
    int row = m0 + (lane & 15);
    int kb  = (lane & 16) ? 8 : 0;
    int KT  = K >> 5;
    const __bf16* Ar = A + (size_t)row * lda;
    const __bf16* Bl = Bp + ((size_t)(n0 >> 4) * KT * 32 + lane) * 16;
    v8f c = {};
    for (int kt = 0; kt < KT; ++kt) {
        Frag a, b;
        a.q[0] = *(const uint4*)(Ar + kt * 32 + kb);
        a.q[1] = *(const uint4*)(Ar + kt * 32 + kb + 16);
        const uint4* bp = (const uint4*)(Bl + (size_t)kt * 512);
        b.q[0] = bp[0];
        b.q[1] = bp[1];
        c = __builtin_amdgcn_wmma_f32_16x16x32_bf16(false, a.v, false, b.v,
                                                    (short)0, c, false, false);
    }
    int n  = n0 + (lane & 15);
    int mb = (lane & 16) ? 8 : 0;
    float bv = bias ? bias[n] : 0.0f;
    float beta = 1.0f, omb = 0.0f;
    if (skipP) { beta = 1.0f / (1.0f + __expf(-skipP[0])); omb = 1.0f - beta; }
#pragma unroll
    for (int i = 0; i < 8; ++i) {
        int m = m0 + mb + i;
        float val = c[i] + bv;
        if (resid) val = beta * val + omb * resid[(size_t)m * N + n];
        if (relu) val = fmaxf(val, 0.0f);
        if (outF) outF[(size_t)m * N + n] = val;
        if (outB) outB[(size_t)m * N + n] = (__bf16)val;
    }
}

// ------------------------- edge attention kernels --------------------------
__global__ void clear_u32_kernel(unsigned* __restrict__ p, long n) {
    long t = (long)blockIdx.x * 256 + threadIdx.x;
    if (t < n) p[t] = 0u;
}

__device__ __forceinline__ float dot32bf(const __bf16* a, const __bf16* b) {
    float acc = 0.0f;
#pragma unroll
    for (int i = 0; i < 4; ++i) {
        union { uint4 u; __bf16 h[8]; } x, y;
        x.u = ((const uint4*)a)[i];
        y.u = ((const uint4*)b)[i];
#pragma unroll
        for (int j = 0; j < 8; ++j) acc += (float)x.h[j] * (float)y.h[j];
    }
    return acc;
}

// a_e = (q[dst,h] . krel[src,h]) * p_rel[h]/sqrt(32); atomicMax encoded into m.
__global__ void attn_score_kernel(const __bf16* __restrict__ kqvDst,
                                  const __bf16* __restrict__ krel,
                                  const int* __restrict__ src, const int* __restrict__ dst,
                                  const float* __restrict__ prel,
                                  float* __restrict__ abuf, unsigned* __restrict__ mEnc,
                                  int E) {
    int t = blockIdx.x * 256 + threadIdx.x;
    if (t >= E * 4) return;
    int e = t >> 2, h = t & 3;
    int si = src[e], di = dst[e];
    const __bf16* q  = kqvDst + (size_t)di * 384 + 128 + h * 32;
    const __bf16* kk = krel + (size_t)si * 128 + h * 32;
    float a = dot32bf(q, kk) * prel[h] * 0.17677669529663687f;
    abuf[t] = a;
    atomicMax(&mEnc[(size_t)di * 4 + h], encF(a));
}

// accumulate sum(e) and sum(e*v) per destination node
__global__ void attn_agg_kernel(const __bf16* __restrict__ vrel,
                                const int* __restrict__ src, const int* __restrict__ dst,
                                const float* __restrict__ abuf,
                                const unsigned* __restrict__ mEnc,
                                float* __restrict__ sSum, float* __restrict__ agg, int E) {
    long t = (long)blockIdx.x * 256 + threadIdx.x;
    if (t >= (long)E * 128) return;
    int e = (int)(t >> 7), c = (int)(t & 127), h = c >> 5;
    int si = src[e], di = dst[e];
    float a  = abuf[(size_t)e * 4 + h];
    float m  = decF(mEnc[(size_t)di * 4 + h]);
    float ex = __expf(a - m);
    float v  = (float)vrel[(size_t)si * 128 + c];
    atomicAdd(&agg[(size_t)di * 128 + c], ex * v);
    if ((c & 31) == 0) atomicAdd(&sSum[(size_t)di * 4 + h], ex);
}

// normalize + gelu -> bf16 activations for the out-projection GEMM
__global__ void attn_fin_kernel(const float* __restrict__ agg, const float* __restrict__ sSum,
                                __bf16* __restrict__ out, int N) {
    int t = blockIdx.x * 256 + threadIdx.x;
    if (t >= N * 128) return;
    int n = t >> 7, h = (t >> 5) & 3;
    float v = agg[t] / (sSum[(size_t)n * 4 + h] + 1e-16f);
    float g = 0.5f * v * (1.0f + tanhf(0.7978845608f * (v + 0.044715f * v * v * v)));
    out[t] = (__bf16)g;
}

// ---------------------- actor head: gathered WMMA GEMM ---------------------
// comb[e] = [x_task[src[e]] | x_proc[dst[e]]]  (K=256), then fused
// relu(comb@W1+b1) . w2 + b2 -> logits. Row sums reduced via LDS ds_add_f32.
__global__ __launch_bounds__(256) void actor_gemm(
    const __bf16* __restrict__ xt, const __bf16* __restrict__ xp,
    const int* __restrict__ src, const int* __restrict__ dst,
    const float* __restrict__ b1, const float* __restrict__ w2,
    const float* __restrict__ b2, const __bf16* __restrict__ Bp,
    float* __restrict__ logits, int E) {
    __shared__ float rowsum[16];
    int wave = threadIdx.x >> 5, lane = threadIdx.x & 31;
    if (threadIdx.x < 16) rowsum[threadIdx.x] = 0.0f;
    __syncthreads();
    int m0 = blockIdx.x << 4;
    int r  = m0 + (lane & 15);
    int si = src[r], di = dst[r];
    int kb = (lane & 16) ? 8 : 0;
    const int KT = 8;  // 256/32
    int n0 = wave << 4;
    const __bf16* Bl = Bp + ((size_t)(n0 >> 4) * KT * 32 + lane) * 16;
    v8f c = {};
#pragma unroll
    for (int kt = 0; kt < KT; ++kt) {
        const __bf16* Arow = (kt < 4) ? (xt + (size_t)si * 128 + kt * 32)
                                      : (xp + (size_t)di * 128 + (kt - 4) * 32);
        Frag a, b;
        a.q[0] = *(const uint4*)(Arow + kb);
        a.q[1] = *(const uint4*)(Arow + kb + 16);
        const uint4* bp = (const uint4*)(Bl + (size_t)kt * 512);
        b.q[0] = bp[0];
        b.q[1] = bp[1];
        c = __builtin_amdgcn_wmma_f32_16x16x32_bf16(false, a.v, false, b.v,
                                                    (short)0, c, false, false);
    }
    int n  = n0 + (lane & 15);
    int mb = (lane & 16) ? 8 : 0;
    float w2n = w2[n], b1n = b1[n];
#pragma unroll
    for (int i = 0; i < 8; ++i) {
        float h = fmaxf(c[i] + b1n, 0.0f);
        atomicAdd(&rowsum[mb + i], h * w2n);
    }
    __syncthreads();
    if (threadIdx.x < 16) logits[m0 + threadIdx.x] = rowsum[threadIdx.x] + b2[0];
}

// ------------------------------ critic head --------------------------------
__global__ void colsum_kernel(const float* __restrict__ X, float* __restrict__ sum, int N) {
    int c  = threadIdx.x;              // 128 threads
    int r0 = blockIdx.x * 128;
    int r1 = min(r0 + 128, N);
    float acc = 0.0f;
    for (int r = r0; r < r1; ++r) acc += X[(size_t)r * 128 + c];
    atomicAdd(&sum[c], acc);
}

__global__ void critic_kernel(const float* __restrict__ csum, const float* __restrict__ w1,
                              const float* __restrict__ b1, const float* __restrict__ w2,
                              const float* __restrict__ b2, float* __restrict__ out, float invN) {
    __shared__ float mean[128];
    __shared__ float red[128];
    int c = threadIdx.x;
    mean[c] = csum[c] * invN;
    __syncthreads();
    float acc = b1[c];
    for (int k = 0; k < 128; ++k) acc += mean[k] * w1[(size_t)k * 128 + c];
    red[c] = fmaxf(acc, 0.0f) * w2[c];
    __syncthreads();
    for (int off = 64; off; off >>= 1) {
        if (c < off) red[c] += red[c + off];
        __syncthreads();
    }
    if (c == 0) out[0] = red[0] + b2[0];
}

// ------------------------------- host driver -------------------------------
extern "C" void kernel_launch(void* const* d_in, const int* in_sizes, int n_in,
                              void* d_out, int out_size, void* d_ws, size_t ws_size,
                              hipStream_t stream) {
    constexpr int Nt = 100000, Np = 10000, Ec = 500000, Ed = 250000;

    const float* x_task  = (const float*)d_in[0];
    const float* x_proc  = (const float*)d_in[1];
    const int*   cro_src = (const int*)d_in[2];
    const int*   cro_dst = (const int*)d_in[3];
    const int*   rev_src = (const int*)d_in[4];
    const int*   rev_dst = (const int*)d_in[5];
    const int*   dep_src = (const int*)d_in[6];
    const int*   dep_dst = (const int*)d_in[7];
    // params pytree flattened in sorted-key order (jax dict flatten):
    const float* actor_b1 = (const float*)d_in[8];
    const float* actor_b2 = (const float*)d_in[9];
    const float* actor_w1 = (const float*)d_in[10];
    const float* actor_w2 = (const float*)d_in[11];
    struct CP {
        const float *krel[3], *vrel[3], *prel[3];
        const float *kqvw[2], *kqvb[2], *outw[2], *outb[2], *skp[2];
    };
    auto getCP = [&](int b) {
        CP p;
        p.krel[0] = (const float*)d_in[b + 0];  p.krel[1] = (const float*)d_in[b + 1];
        p.krel[2] = (const float*)d_in[b + 2];
        p.kqvb[1] = (const float*)d_in[b + 3];  p.kqvb[0] = (const float*)d_in[b + 4];
        p.kqvw[1] = (const float*)d_in[b + 5];  p.kqvw[0] = (const float*)d_in[b + 6];
        p.outb[1] = (const float*)d_in[b + 7];  p.outb[0] = (const float*)d_in[b + 8];
        p.outw[1] = (const float*)d_in[b + 9];  p.outw[0] = (const float*)d_in[b + 10];
        p.prel[0] = (const float*)d_in[b + 11]; p.prel[1] = (const float*)d_in[b + 12];
        p.prel[2] = (const float*)d_in[b + 13];
        p.skp[1]  = (const float*)d_in[b + 14]; p.skp[0]  = (const float*)d_in[b + 15];
        p.vrel[0] = (const float*)d_in[b + 16]; p.vrel[1] = (const float*)d_in[b + 17];
        p.vrel[2] = (const float*)d_in[b + 18];
        return p;
    };
    CP cp[2] = {getCP(12), getCP(31)};
    const float* critic_b1  = (const float*)d_in[50];
    const float* critic_b2  = (const float*)d_in[51];
    const float* critic_w1  = (const float*)d_in[52];
    const float* critic_w2  = (const float*)d_in[53];
    const float* proc_emb_b = (const float*)d_in[54];
    const float* proc_emb_w = (const float*)d_in[55];
    const float* task_emb_b = (const float*)d_in[56];
    const float* task_emb_w = (const float*)d_in[57];

    // -------- workspace layout (bump allocator, 256B aligned) --------
    char* ws = (char*)d_ws;
    size_t off = 0;
    auto alloc = [&](size_t bytes) -> char* {
        char* r = ws + off;
        off = (off + bytes + 255) & ~(size_t)255;
        return r;
    };
    float*  curT = (float*)alloc((size_t)Nt * 128 * 4);
    float*  curP = (float*)alloc((size_t)Np * 128 * 4);
    float*  nxtT = (float*)alloc((size_t)Nt * 128 * 4);
    float*  nxtP = (float*)alloc((size_t)Np * 128 * 4);
    __bf16* actT = (__bf16*)alloc((size_t)Nt * 128 * 2);
    __bf16* actP = (__bf16*)alloc((size_t)Np * 128 * 2);
    __bf16* kqvT = (__bf16*)alloc((size_t)Nt * 384 * 2);
    __bf16* kqvP = (__bf16*)alloc((size_t)Np * 384 * 2);
    __bf16* kN[3], *vN[3];  // per-relation per-node transformed K/V
    kN[0] = (__bf16*)alloc((size_t)Nt * 128 * 2);
    vN[0] = (__bf16*)alloc((size_t)Nt * 128 * 2);
    kN[1] = (__bf16*)alloc((size_t)Nt * 128 * 2);
    vN[1] = (__bf16*)alloc((size_t)Nt * 128 * 2);
    kN[2] = (__bf16*)alloc((size_t)Np * 128 * 2);
    vN[2] = (__bf16*)alloc((size_t)Np * 128 * 2);
    unsigned* mT = (unsigned*)alloc((size_t)Nt * 4 * 4);
    unsigned* mP = (unsigned*)alloc((size_t)Np * 4 * 4);
    float* sT   = (float*)alloc((size_t)Nt * 4 * 4);
    float* sP   = (float*)alloc((size_t)Np * 4 * 4);
    float* aggT = (float*)alloc((size_t)Nt * 128 * 4);
    float* aggP = (float*)alloc((size_t)Np * 128 * 4);
    float* ab0  = (float*)alloc((size_t)Ec * 4 * 4);
    float* ab1  = (float*)alloc((size_t)Ed * 4 * 4);
    float* ab2  = (float*)alloc((size_t)Ec * 4 * 4);
    __bf16* gactT = (__bf16*)alloc((size_t)Nt * 128 * 2);
    __bf16* gactP = (__bf16*)alloc((size_t)Np * 128 * 2);
    __bf16* pkKqv[2][2], *pkOut[2][2], *pkK[2][3], *pkV[2][3];
    for (int l = 0; l < 2; ++l) {
        for (int t = 0; t < 2; ++t) pkKqv[l][t] = (__bf16*)alloc(128 * 384 * 2);
        for (int t = 0; t < 2; ++t) pkOut[l][t] = (__bf16*)alloc(128 * 128 * 2);
        for (int r = 0; r < 3; ++r) pkK[l][r]   = (__bf16*)alloc(128 * 128 * 2);
        for (int r = 0; r < 3; ++r) pkV[l][r]   = (__bf16*)alloc(128 * 128 * 2);
    }
    __bf16* pkActor = (__bf16*)alloc(256 * 128 * 2);
    float*  csum    = (float*)alloc(128 * 4);
    (void)ws_size; (void)in_sizes; (void)n_in; (void)out_size;

    auto pack = [&](const float* W, __bf16* dst, int K, int N, int mode) {
        pack_b_kernel<<<(K * N + 255) / 256, 256, 0, stream>>>(W, dst, K, N, mode);
    };
    // ---- pre-swizzle all weights into WMMA B layout ----
    for (int l = 0; l < 2; ++l) {
        pack(cp[l].kqvw[0], pkKqv[l][0], 128, 384, 0);
        pack(cp[l].kqvw[1], pkKqv[l][1], 128, 384, 0);
        pack(cp[l].outw[0], pkOut[l][0], 128, 128, 0);
        pack(cp[l].outw[1], pkOut[l][1], 128, 128, 0);
        for (int r = 0; r < 3; ++r) {
            pack(cp[l].krel[r], pkK[l][r], 128, 128, 1);  // block-diagonal
            pack(cp[l].vrel[r], pkV[l][r], 128, 128, 1);
        }
    }
    pack(actor_w1, pkActor, 256, 128, 0);

    // ---- embeddings ----
    embed_kernel<<<(Nt * 128 + 255) / 256, 256, 0, stream>>>(
        x_task, task_emb_w, task_emb_b, curT, actT, Nt, 6);
    embed_kernel<<<(Np * 128 + 255) / 256, 256, 0, stream>>>(
        x_proc, proc_emb_w, proc_emb_b, curP, actP, Np, 4);

    auto G = [&](const __bf16* A, int lda, const __bf16* Bp, const float* bias,
                 const float* resid, const float* skp, float* oF, __bf16* oB,
                 int M, int K, int N, int relu) {
        dim3 grid(M / 16, N / 128);
        gemm_wmma<<<grid, 256, 0, stream>>>(A, lda, Bp, bias, resid, skp, oF, oB,
                                            M, K, N, relu);
    };
    auto clr = [&](void* p, long n32) {
        clear_u32_kernel<<<(unsigned)((n32 + 255) / 256), 256, 0, stream>>>(
            (unsigned*)p, n32);
    };

    float *cT = curT, *cP = curP, *nT = nxtT, *nP = nxtP;
    const int* eS[3] = {cro_src, dep_src, rev_src};
    const int* eD[3] = {cro_dst, dep_dst, rev_dst};
    const int  eE[3] = {Ec, Ed, Ec};
    float*     eA[3] = {ab0, ab1, ab2};

    for (int l = 0; l < 2; ++l) {
        // KQV projections  [N x 128] @ [128 x 384]
        G(actT, 128, pkKqv[l][0], cp[l].kqvb[0], nullptr, nullptr, nullptr, kqvT,
          Nt, 128, 384, 0);
        G(actP, 128, pkKqv[l][1], cp[l].kqvb[1], nullptr, nullptr, nullptr, kqvP,
          Np, 128, 384, 0);
        // per-relation per-node K/V transforms (block-diag over heads)
        G(kqvT, 384, pkK[l][0], nullptr, nullptr, nullptr, nullptr, kN[0], Nt, 128, 128, 0);
        G(kqvT + 256, 384, pkV[l][0], nullptr, nullptr, nullptr, nullptr, vN[0], Nt, 128, 128, 0);
        G(kqvT, 384, pkK[l][1], nullptr, nullptr, nullptr, nullptr, kN[1], Nt, 128, 128, 0);
        G(kqvT + 256, 384, pkV[l][1], nullptr, nullptr, nullptr, nullptr, vN[1], Nt, 128, 128, 0);
        G(kqvP, 384, pkK[l][2], nullptr, nullptr, nullptr, nullptr, kN[2], Np, 128, 128, 0);
        G(kqvP + 256, 384, pkV[l][2], nullptr, nullptr, nullptr, nullptr, vN[2], Np, 128, 128, 0);
        // segment softmax state
        clr(mT, (long)Nt * 4);  clr(mP, (long)Np * 4);
        clr(sT, (long)Nt * 4);  clr(sP, (long)Np * 4);
        clr(aggT, (long)Nt * 128); clr(aggP, (long)Np * 128);
        // scores + running max (rel0 -> proc, rel1/rel2 -> task)
        const __bf16* dstKqv[3] = {kqvP, kqvT, kqvT};
        unsigned*     dstM[3]   = {mP, mT, mT};
        for (int r = 0; r < 3; ++r)
            attn_score_kernel<<<(eE[r] * 4 + 255) / 256, 256, 0, stream>>>(
                dstKqv[r], kN[r], eS[r], eD[r], cp[l].prel[r], eA[r], dstM[r], eE[r]);
        float* dstS[3] = {sP, sT, sT};
        float* dstA[3] = {aggP, aggT, aggT};
        for (int r = 0; r < 3; ++r)
            attn_agg_kernel<<<(unsigned)(((long)eE[r] * 128 + 255) / 256), 256, 0, stream>>>(
                vN[r], eS[r], eD[r], eA[r], dstM[r], dstS[r], dstA[r], eE[r]);
        // normalize + gelu
        attn_fin_kernel<<<(Nt * 128 + 255) / 256, 256, 0, stream>>>(aggT, sT, gactT, Nt);
        attn_fin_kernel<<<(Np * 128 + 255) / 256, 256, 0, stream>>>(aggP, sP, gactP, Np);
        // out projection + learned skip (+relu between layers)
        int relu = (l == 0) ? 1 : 0;
        G(gactT, 128, pkOut[l][0], cp[l].outb[0], cT, cp[l].skp[0], nT, actT,
          Nt, 128, 128, relu);
        G(gactP, 128, pkOut[l][1], cp[l].outb[1], cP, cp[l].skp[1], nP, actP,
          Np, 128, 128, relu);
        float* t;
        t = cT; cT = nT; nT = t;
        t = cP; cP = nP; nP = t;
    }

    // ---- actor head: gathered GEMM, fused relu + dot(w2) ----
    float* outF = (float*)d_out;
    actor_gemm<<<Ec / 16, 256, 0, stream>>>(actT, actP, cro_src, cro_dst,
                                            actor_b1, actor_w2, actor_b2,
                                            pkActor, outF, Ec);
    // ---- critic head ----
    clr(csum, 128);
    colsum_kernel<<<(Nt + 127) / 128, 128, 0, stream>>>(cT, csum, Nt);
    critic_kernel<<<1, 128, 0, stream>>>(csum, critic_w1, critic_b1, critic_w2,
                                         critic_b2, outF + Ec, 1.0f / Nt);
}
